// LinearAttention_76020921139201
// MI455X (gfx1250) — matrix operand order repaired and checked by reading, not compiled, not run on hardware
//
#include <hip/hip_runtime.h>
#include <cstdint>
#include <cstddef>

// ---------------------------------------------------------------------------
// Problem constants (from reference): B=4, T=8192, D=1024, H=16, hd=64
// ---------------------------------------------------------------------------
#define BATCH 4
#define SEQT  8192
#define DIMD  1024
#define NHEAD 16
#define HDIM  64
#define MTOT  (BATCH * SEQT)   // 32768 rows

typedef __attribute__((ext_vector_type(16))) __bf16 bf16x16;
typedef __attribute__((ext_vector_type(8)))  float  f32x8;

union FragU { bf16x16 v; uint4 q[2]; };

__device__ __forceinline__ unsigned short f2bf(float f) {
  union { float f; uint32_t u; } c; c.f = f;
  uint32_t u = c.u;
  u += 0x7FFFu + ((u >> 16) & 1u);           // round-to-nearest-even
  return (unsigned short)(u >> 16);
}
__device__ __forceinline__ float bf2f(unsigned short h) {
  union { uint32_t u; float f; } c; c.u = ((uint32_t)h) << 16; return c.f;
}

// ---------------------------------------------------------------------------
// CDNA5 async global->LDS copy (ASYNCcnt-tracked).  Generic LDS pointer's low
// 32 bits are the LDS byte offset.
// ---------------------------------------------------------------------------
__device__ __forceinline__ void async_copy_b128(const unsigned short* gsrc,
                                                unsigned short* lds_dst) {
  uint32_t laddr = (uint32_t)(uintptr_t)lds_dst;
  asm volatile("global_load_async_to_lds_b128 %0, %1, off"
               :: "v"(laddr), "v"(gsrc) : "memory");
}
__device__ __forceinline__ void wait_async_le2() {
  asm volatile("s_wait_asynccnt 0x2" ::: "memory");
}
__device__ __forceinline__ void wait_async_le0() {
  asm volatile("s_wait_asynccnt 0x0" ::: "memory");
}

// A fragment (16x32 bf16) from row-major A[M,K]:
// lane L -> row m0+(L&15); 16 contiguous K at k0 + (L>=16 ? 16 : 0)
__device__ __forceinline__ bf16x16 fragA_g(const unsigned short* base, int ld,
                                           int row, int k0, int lane) {
  FragU f;
  const unsigned short* p = base + (size_t)row * ld + k0 + ((lane >> 4) << 4);
  f.q[0] = *(const uint4*)(p);
  f.q[1] = *(const uint4*)(p + 8);
  return f.v;
}
// B fragment (32x16 bf16) from column-accessible Bs[n*ld + k] (= W row-major):
// lane L -> col n0+(L&15); 16 contiguous K at k0 + (L>=16 ? 16 : 0)
__device__ __forceinline__ bf16x16 fragB_g(const unsigned short* base, int ld,
                                           int col, int k0, int lane) {
  FragU f;
  const unsigned short* p = base + (size_t)col * ld + k0 + ((lane >> 4) << 4);
  f.q[0] = *(const uint4*)(p);
  f.q[1] = *(const uint4*)(p + 8);
  return f.v;
}
// B fragment from LDS tile Bs[col][k], pitch BPITCH elements (conflict-free).
#define BPITCH 72
__device__ __forceinline__ bf16x16 fragB_lds(const unsigned short* bs, int col,
                                             int k32, int lane) {
  FragU f;
  const unsigned short* p = bs + col * BPITCH + k32 + ((lane >> 4) << 4);
  f.q[0] = *(const uint4*)(p);
  f.q[1] = *(const uint4*)(p + 8);
  return f.v;
}

__device__ __forceinline__ f32x8 wmma_bf16(bf16x16 a, bf16x16 b, f32x8 c) {
  return __builtin_amdgcn_wmma_f32_16x16x32_bf16(false, a, false, b, (short)0,
                                                 c, false, false);
}

// ---------------------------------------------------------------------------
// Elementwise helpers
// ---------------------------------------------------------------------------
__global__ void cvt_f32_bf16_kernel(const float* __restrict__ src,
                                    unsigned short* __restrict__ dst, int n) {
  int g = blockIdx.x * blockDim.x + threadIdx.x;
  if (g < n) dst[g] = f2bf(src[g]);
}

__global__ void zero_f32_kernel(float* __restrict__ p, int n) {
  int g = blockIdx.x * blockDim.x + threadIdx.x;
  if (g < n) p[g] = 0.0f;
}

// ---------------------------------------------------------------------------
// Fused QKV projection: C = x @ W^T + b, optional feature map elu(x)+1.
// grid: (MTOT/256, DIMD/64, 3), block: 256 (8 waves), wave -> 32x64 strip
// (2 A-frags x 4 shared B-frags = 8 WMMA per 32-k step).
// B tile (64 cols x 64 k) double-buffered in LDS via async global->LDS.
// ---------------------------------------------------------------------------
__global__ __launch_bounds__(256) void qkv_gemm_kernel(
    const unsigned short* __restrict__ xb,
    const unsigned short* __restrict__ wqb,
    const unsigned short* __restrict__ wkb,
    const unsigned short* __restrict__ wvb,
    const float* __restrict__ bq, const float* __restrict__ bk,
    const float* __restrict__ bv,
    unsigned short* __restrict__ qo, unsigned short* __restrict__ ko,
    unsigned short* __restrict__ vo) {
  __shared__ unsigned short Bs[2][64 * BPITCH];

  const int which = blockIdx.z;
  const unsigned short* w = (which == 0) ? wqb : (which == 1) ? wkb : wvb;
  const float* bias       = (which == 0) ? bq  : (which == 1) ? bk  : bv;
  unsigned short* out     = (which == 0) ? qo  : (which == 1) ? ko  : vo;
  const bool act = (which < 2);

  const int wave = threadIdx.x >> 5, lane = threadIdx.x & 31;
  const int m0 = blockIdx.x * 256 + wave * 32;
  const int n0 = blockIdx.y * 64;
  const int rowA = m0 + (lane & 15);
  const int c0 = threadIdx.x * 2;          // 512 16B-chunks per tile, 2/thread

  f32x8 acc[2][4];
#pragma unroll
  for (int i = 0; i < 2; ++i)
#pragma unroll
    for (int j = 0; j < 4; ++j)
#pragma unroll
      for (int r = 0; r < 8; ++r) acc[i][j][r] = 0.0f;

  auto prefetchB = [&](int buf, int k0) {
#pragma unroll
    for (int t = 0; t < 2; ++t) {
      const int c = c0 + t;
      const int col = c >> 3, kq = (c & 7) * 8;
      async_copy_b128(w + (size_t)(n0 + col) * DIMD + k0 + kq,
                      &Bs[buf][col * BPITCH + kq]);
    }
  };

  prefetchB(0, 0);
  for (int kt = 0; kt < DIMD / 64; ++kt) {
    const int k0 = kt * 64;
    if (kt + 1 < DIMD / 64) {
      prefetchB((kt + 1) & 1, k0 + 64);
      wait_async_le2();                    // current tile's 2 ops retired
    } else {
      wait_async_le0();
    }
    __syncthreads();
    const unsigned short* bs = Bs[kt & 1];
#pragma unroll
    for (int kk = 0; kk < 2; ++kk) {
      bf16x16 a0 = fragA_g(xb, DIMD, rowA, k0 + kk * 32, lane);
      bf16x16 a1 = fragA_g(xb, DIMD, rowA + 16, k0 + kk * 32, lane);
      bf16x16 bq_[4];
#pragma unroll
      for (int j = 0; j < 4; ++j)
        bq_[j] = fragB_lds(bs, j * 16 + (lane & 15), kk * 32, lane);
#pragma unroll
      for (int j = 0; j < 4; ++j) acc[0][j] = wmma_bf16(a0, bq_[j], acc[0][j]);
#pragma unroll
      for (int j = 0; j < 4; ++j) acc[1][j] = wmma_bf16(a1, bq_[j], acc[1][j]);
    }
    __syncthreads();                       // tile consumed before overwrite
  }

#pragma unroll
  for (int i = 0; i < 2; ++i)
#pragma unroll
    for (int j = 0; j < 4; ++j) {
      const int col = n0 + j * 16 + (lane & 15);
      const float bsv = bias[col];
#pragma unroll
      for (int r = 0; r < 8; ++r) {
        const int row = m0 + i * 16 + r + ((lane >> 4) << 3);
        float v = acc[i][j][r] + bsv;
        if (act) v = (v > 0.0f) ? (v + 1.0f) : __expf(v);  // elu(v)+1
        out[(size_t)row * DIMD + col] = f2bf(v);
      }
    }
}

// ---------------------------------------------------------------------------
// kv[b,h,m,d] += sum_t v[t,m] * k[t,d] over a T-chunk.
// grid: (B*H, SEQT/1024), block 256. LDS-transposed k,v tiles (32 t x 64 d),
// row pitch 40 bf16 (20 banks) -> conflict-free ds_load_b128 fragments.
// ---------------------------------------------------------------------------
#define KV_PITCH 40
__global__ __launch_bounds__(256) void kv_accum_kernel(
    const unsigned short* __restrict__ kb, const unsigned short* __restrict__ vb,
    float* __restrict__ kvg) {
  __shared__ unsigned short ldsK[HDIM * KV_PITCH];
  __shared__ unsigned short ldsV[HDIM * KV_PITCH];

  const int bh = blockIdx.x;
  const int b = bh >> 4, h = bh & 15;
  const int wave = threadIdx.x >> 5, lane = threadIdx.x & 31;
  const int tl = threadIdx.x >> 3;        // 0..31: local t row to stage
  const int dg = (threadIdx.x & 7) * 8;   // 0..56: d group of 8

  f32x8 acc[2];
#pragma unroll
  for (int p = 0; p < 2; ++p)
#pragma unroll
    for (int r = 0; r < 8; ++r) acc[p][r] = 0.0f;

  const int t0base = blockIdx.y * 1024;
  for (int t0 = t0base; t0 < t0base + 1024; t0 += 32) {
    const size_t g = ((size_t)(b * SEQT + t0 + tl)) * DIMD + h * HDIM + dg;
    union { uint4 q; unsigned short s[8]; } uk, uv;
    uk.q = *(const uint4*)(kb + g);
    uv.q = *(const uint4*)(vb + g);
    __syncthreads();                       // prior fragment reads done
#pragma unroll
    for (int e = 0; e < 8; ++e) {
      ldsK[(dg + e) * KV_PITCH + tl] = uk.s[e];   // transpose: [d][t]
      ldsV[(dg + e) * KV_PITCH + tl] = uv.s[e];
    }
    __syncthreads();
    FragU fa[2], fb[2];
#pragma unroll
    for (int p = 0; p < 2; ++p) {
      const int tid = wave * 2 + p;        // 8 waves x 2 = 16 tiles (4x4)
      const int mi = tid >> 2, di = tid & 3;
      {  // A = v^T : row m, contiguous t
        const unsigned short* pa =
            &ldsV[(mi * 16 + (lane & 15)) * KV_PITCH + ((lane >> 4) << 4)];
        fa[p].q[0] = *(const uint4*)(pa);
        fa[p].q[1] = *(const uint4*)(pa + 8);
      }
      {  // B = k : col d, contiguous t
        const unsigned short* pb =
            &ldsK[(di * 16 + (lane & 15)) * KV_PITCH + ((lane >> 4) << 4)];
        fb[p].q[0] = *(const uint4*)(pb);
        fb[p].q[1] = *(const uint4*)(pb + 8);
      }
    }
#pragma unroll
    for (int p = 0; p < 2; ++p)
      acc[p] = wmma_bf16(fa[p].v, fb[p].v, acc[p]);
  }

  float* kvdst = kvg + (size_t)bh * HDIM * HDIM;
#pragma unroll
  for (int p = 0; p < 2; ++p) {
    const int tid = wave * 2 + p;
    const int mi = tid >> 2, di = tid & 3;
#pragma unroll
    for (int r = 0; r < 8; ++r) {
      const int m = mi * 16 + r + ((lane >> 4) << 3);
      const int d = di * 16 + (lane & 15);
      atomicAdd(&kvdst[m * HDIM + d], acc[p][r]);
    }
  }
}

// k_sum[b,h,d] = sum_t k[b,t,h,d].  grid (B*H, 8), block 256.
__global__ void ksum_kernel(const unsigned short* __restrict__ kb,
                            float* __restrict__ ksum) {
  const int bh = blockIdx.x;
  const int b = bh >> 4, h = bh & 15;
  const int d = threadIdx.x & 63;
  const int sub = threadIdx.x >> 6;            // 0..3
  const int t0 = blockIdx.y * 1024 + sub * 256;
  float s = 0.0f;
  for (int t = 0; t < 256; ++t)
    s += bf2f(kb[((size_t)(b * SEQT + t0 + t)) * DIMD + h * HDIM + d]);
  atomicAdd(&ksum[(size_t)bh * HDIM + d], s);
}

// z[row,h] = 1 / (sum_d q[row, h*64+d] * ksum[b,h,d] + eps)
__global__ void z_kernel(const unsigned short* __restrict__ qb,
                         const float* __restrict__ ksum,
                         float* __restrict__ zbuf) {
  const int g = blockIdx.x * blockDim.x + threadIdx.x;  // < MTOT*NHEAD
  const int row = g >> 4, h = g & 15;
  const int b = row / SEQT;
  const float* ks = ksum + ((size_t)b * NHEAD + h) * HDIM;
  const unsigned short* qp = qb + (size_t)row * DIMD + h * HDIM;
  float s = 0.0f;
#pragma unroll 8
  for (int d = 0; d < HDIM; ++d) s += bf2f(qp[d]) * ks[d];
  zbuf[g] = 1.0f / (s + 1e-6f);
}

// y[row, h*64+m] = z * sum_d q[row,h*64+d] * kv[b,h,m,d]
// grid (MTOT/256, NHEAD), block 256 (8 waves, 32x64 per wave).
__global__ __launch_bounds__(256) void attn_apply_kernel(
    const unsigned short* __restrict__ qb,
    const unsigned short* __restrict__ kvb,
    const float* __restrict__ zbuf, unsigned short* __restrict__ yb) {
  const int wave = threadIdx.x >> 5, lane = threadIdx.x & 31;
  const int h = blockIdx.y;
  const int m0 = blockIdx.x * 256 + wave * 32;
  const int b = (blockIdx.x * 256) / SEQT;   // 256-row blocks never cross b
  const unsigned short* kvh = kvb + ((size_t)(b * NHEAD + h)) * HDIM * HDIM;
  const int rowA = m0 + (lane & 15);

  f32x8 acc[2][4];
#pragma unroll
  for (int i = 0; i < 2; ++i)
#pragma unroll
    for (int j = 0; j < 4; ++j)
#pragma unroll
      for (int r = 0; r < 8; ++r) acc[i][j][r] = 0.0f;

#pragma unroll
  for (int k0 = 0; k0 < HDIM; k0 += 32) {
    bf16x16 a0 = fragA_g(qb, DIMD, rowA, h * HDIM + k0, lane);
    bf16x16 a1 = fragA_g(qb, DIMD, rowA + 16, h * HDIM + k0, lane);
    bf16x16 bq_[4];
#pragma unroll
    for (int j = 0; j < 4; ++j)
      bq_[j] = fragB_g(kvh, HDIM, j * 16 + (lane & 15), k0, lane);
#pragma unroll
    for (int j = 0; j < 4; ++j) acc[0][j] = wmma_bf16(a0, bq_[j], acc[0][j]);
#pragma unroll
    for (int j = 0; j < 4; ++j) acc[1][j] = wmma_bf16(a1, bq_[j], acc[1][j]);
  }

#pragma unroll
  for (int i = 0; i < 2; ++i)
#pragma unroll
    for (int j = 0; j < 4; ++j) {
      const int col = j * 16 + (lane & 15);
#pragma unroll
      for (int r = 0; r < 8; ++r) {
        const int row = m0 + i * 16 + r + ((lane >> 4) << 3);
        const float zz = zbuf[(size_t)row * NHEAD + h];
        yb[(size_t)row * DIMD + h * HDIM + col] = f2bf(acc[i][j][r] * zz);
      }
    }
}

// out = y @ Wo^T + bo (f32 output). grid (MTOT/256, DIMD/64), block 256.
// Same async-LDS double-buffered B staging + 32x64 per-wave tile.
__global__ __launch_bounds__(256) void out_gemm_kernel(
    const unsigned short* __restrict__ yb, const unsigned short* __restrict__ wob,
    const float* __restrict__ bo, float* __restrict__ out) {
  __shared__ unsigned short Bs[2][64 * BPITCH];

  const int wave = threadIdx.x >> 5, lane = threadIdx.x & 31;
  const int m0 = blockIdx.x * 256 + wave * 32;
  const int n0 = blockIdx.y * 64;
  const int rowA = m0 + (lane & 15);
  const int c0 = threadIdx.x * 2;

  f32x8 acc[2][4];
#pragma unroll
  for (int i = 0; i < 2; ++i)
#pragma unroll
    for (int j = 0; j < 4; ++j)
#pragma unroll
      for (int r = 0; r < 8; ++r) acc[i][j][r] = 0.0f;

  auto prefetchB = [&](int buf, int k0) {
#pragma unroll
    for (int t = 0; t < 2; ++t) {
      const int c = c0 + t;
      const int col = c >> 3, kq = (c & 7) * 8;
      async_copy_b128(wob + (size_t)(n0 + col) * DIMD + k0 + kq,
                      &Bs[buf][col * BPITCH + kq]);
    }
  };

  prefetchB(0, 0);
  for (int kt = 0; kt < DIMD / 64; ++kt) {
    const int k0 = kt * 64;
    if (kt + 1 < DIMD / 64) {
      prefetchB((kt + 1) & 1, k0 + 64);
      wait_async_le2();
    } else {
      wait_async_le0();
    }
    __syncthreads();
    const unsigned short* bs = Bs[kt & 1];
#pragma unroll
    for (int kk = 0; kk < 2; ++kk) {
      bf16x16 a0 = fragA_g(yb, DIMD, rowA, k0 + kk * 32, lane);
      bf16x16 a1 = fragA_g(yb, DIMD, rowA + 16, k0 + kk * 32, lane);
      bf16x16 bq_[4];
#pragma unroll
      for (int j = 0; j < 4; ++j)
        bq_[j] = fragB_lds(bs, j * 16 + (lane & 15), kk * 32, lane);
#pragma unroll
      for (int j = 0; j < 4; ++j) acc[0][j] = wmma_bf16(a0, bq_[j], acc[0][j]);
#pragma unroll
      for (int j = 0; j < 4; ++j) acc[1][j] = wmma_bf16(a1, bq_[j], acc[1][j]);
    }
    __syncthreads();
  }

#pragma unroll
  for (int i = 0; i < 2; ++i)
#pragma unroll
    for (int j = 0; j < 4; ++j) {
      const int col = n0 + j * 16 + (lane & 15);
      const float bsv = bo[col];
#pragma unroll
      for (int r = 0; r < 8; ++r) {
        const int row = m0 + i * 16 + r + ((lane >> 4) << 3);
        out[(size_t)row * DIMD + col] = acc[i][j][r] + bsv;
      }
    }
}

// ---------------------------------------------------------------------------
// Host orchestration
// ---------------------------------------------------------------------------
extern "C" void kernel_launch(void* const* d_in, const int* in_sizes, int n_in,
                              void* d_out, int out_size, void* d_ws,
                              size_t ws_size, hipStream_t stream) {
  (void)in_sizes; (void)n_in; (void)out_size; (void)ws_size;
  const float* x  = (const float*)d_in[0];
  const float* Wq = (const float*)d_in[1];
  const float* bq = (const float*)d_in[2];
  const float* Wk = (const float*)d_in[3];
  const float* bk = (const float*)d_in[4];
  const float* Wv = (const float*)d_in[5];
  const float* bv = (const float*)d_in[6];
  const float* Wo = (const float*)d_in[7];
  const float* bo = (const float*)d_in[8];
  float* out = (float*)d_out;

  char* ws = (char*)d_ws;
  size_t off = 0;
  auto take = [&](size_t bytes) -> char* {
    char* p = ws + off;
    off += (bytes + 255) & ~(size_t)255;
    return p;
  };
  const size_t mat_b = (size_t)MTOT * DIMD * sizeof(unsigned short);  // 64 MB
  unsigned short* xb  = (unsigned short*)take(mat_b);  // reused as yb later
  unsigned short* qb  = (unsigned short*)take(mat_b);
  unsigned short* kb  = (unsigned short*)take(mat_b);
  unsigned short* vb  = (unsigned short*)take(mat_b);
  unsigned short* wqb = (unsigned short*)take((size_t)DIMD * DIMD * 2);
  unsigned short* wkb = (unsigned short*)take((size_t)DIMD * DIMD * 2);
  unsigned short* wvb = (unsigned short*)take((size_t)DIMD * DIMD * 2);
  unsigned short* wob = (unsigned short*)take((size_t)DIMD * DIMD * 2);
  float* kvg = (float*)take((size_t)BATCH * NHEAD * HDIM * HDIM * 4);
  unsigned short* kvb = (unsigned short*)take((size_t)BATCH * NHEAD * HDIM * HDIM * 2);
  float* ksum = (float*)take((size_t)BATCH * NHEAD * HDIM * 4);
  float* zbuf = (float*)take((size_t)MTOT * NHEAD * 4);
  unsigned short* yb = xb;  // x dead after QKV GEMM

  const int nX = MTOT * DIMD;         // 33,554,432
  const int nW = DIMD * DIMD;         // 1,048,576
  const int nKV = BATCH * NHEAD * HDIM * HDIM;  // 262,144
  const int nKS = BATCH * NHEAD * HDIM;         // 4,096

  // 1) convert inputs to bf16
  cvt_f32_bf16_kernel<<<(nX + 255) / 256, 256, 0, stream>>>(x, xb, nX);
  cvt_f32_bf16_kernel<<<(nW + 255) / 256, 256, 0, stream>>>(Wq, wqb, nW);
  cvt_f32_bf16_kernel<<<(nW + 255) / 256, 256, 0, stream>>>(Wk, wkb, nW);
  cvt_f32_bf16_kernel<<<(nW + 255) / 256, 256, 0, stream>>>(Wv, wvb, nW);
  cvt_f32_bf16_kernel<<<(nW + 255) / 256, 256, 0, stream>>>(Wo, wob, nW);

  // 2) zero accumulators (deterministic across graph replays)
  zero_f32_kernel<<<(nKV + 255) / 256, 256, 0, stream>>>(kvg, nKV);
  zero_f32_kernel<<<(nKS + 255) / 256, 256, 0, stream>>>(ksum, nKS);

  // 3) fused QKV GEMM (+elu+1 on q,k)
  qkv_gemm_kernel<<<dim3(MTOT / 256, DIMD / 64, 3), 256, 0, stream>>>(
      xb, wqb, wkb, wvb, bq, bk, bv, qb, kb, vb);

  // 4) k_sum and kv accumulation
  ksum_kernel<<<dim3(BATCH * NHEAD, SEQT / 1024), 256, 0, stream>>>(kb, ksum);
  kv_accum_kernel<<<dim3(BATCH * NHEAD, SEQT / 1024), 256, 0, stream>>>(kb, vb, kvg);
  cvt_f32_bf16_kernel<<<(nKV + 255) / 256, 256, 0, stream>>>(kvg, kvb, nKV);

  // 5) normalizer
  z_kernel<<<(MTOT * NHEAD) / 256, 256, 0, stream>>>(qb, ksum, zbuf);

  // 6) y = (q @ kv) * z
  attn_apply_kernel<<<dim3(MTOT / 256, NHEAD), 256, 0, stream>>>(qb, kvb, zbuf, yb);

  // 7) out = y @ Wo^T + bo
  out_gemm_kernel<<<dim3(MTOT / 256, DIMD / 64), 256, 0, stream>>>(yb, wob, bo, out);
}
